// partical_similarity_71193377898738
// MI455X (gfx1250) — matrix operand order repaired and checked by reading, not compiled, not run on hardware
//
#include <hip/hip_runtime.h>
#include <stdint.h>

// ---------------- types ----------------
typedef unsigned int  u32x4 __attribute__((ext_vector_type(4)));
typedef float         v8f   __attribute__((ext_vector_type(8)));
typedef __bf16        v16bf __attribute__((ext_vector_type(16)));

typedef unsigned int  tdm_u4 __attribute__((ext_vector_type(4)));
typedef int           tdm_i8 __attribute__((ext_vector_type(8)));
typedef int           tdm_i4 __attribute__((ext_vector_type(4)));

struct Frag32 { u32x4 lo, hi; };                    // 32 bytes = 16 bf16
struct __attribute__((packed)) U4U { u32x4 v; };    // unaligned 16B load

static __device__ __forceinline__ u32x4 ldg_u4_unaligned(const unsigned short* p) {
  return ((const U4U*)p)->v;
}

// ---------------- problem constants ----------------
#define NB   16      // batch (both grd samples 'o' and sat samples 'n')
#define CCH  32      // channels
#define HK   32      // kernel h
#define WK   32      // kernel w
#define HS   96      // sat h
#define WS   96      // sat w
#define HP   128     // padded h
#define WP   128     // padded w
#define HO   97      // out h
#define WO   97      // out w
#define KTOT 32768   // C*HK*WK
#define TXN  7       // x tiles of 16 (covers 97)
#define YPR  49      // y tile-pairs (2 rows each, covers 97)
#define T2   (YPR*TXN)          // 343 tile-pairs per sat sample
#define TOT2 (NB*T2)            // 5488 waves total
#define KC   1024    // B panel chunk (= one channel) staged in LDS: 16*KC bf16 = 32 KB
#define EPSV 1e-12f

#define HAVE_TDM __has_builtin(__builtin_amdgcn_tensor_load_to_lds)

// ---------------- helpers ----------------
static __device__ __forceinline__ float rinv_norm(float ss) {
  return 1.0f / fmaxf(sqrtf(ss), EPSV);
}
static __device__ __forceinline__ unsigned short f2bf(float f) {
  uint32_t u = __float_as_uint(f);
  u += 0x7FFFu + ((u >> 16) & 1u);   // round to nearest even
  return (unsigned short)(u >> 16);
}

// ---------------- 1) per-sample sum of squares ----------------
__global__ __launch_bounds__(256)
void k_sumsq(const float* __restrict__ x, float* __restrict__ out,
             int blocksPerSample, int perSample) {
  int n     = blockIdx.x / blocksPerSample;
  int local = (blockIdx.x % blocksPerSample) * 256 + threadIdx.x;
  float v = x[(size_t)n * perSample + local];
  v *= v;
  #pragma unroll
  for (int off = 16; off; off >>= 1) v += __shfl_down(v, off, 32);
  __shared__ float sred[8];
  if ((threadIdx.x & 31) == 0) sred[threadIdx.x >> 5] = v;
  __syncthreads();
  if (threadIdx.x == 0) {
    float s = 0.f;
    #pragma unroll
    for (int i = 0; i < 8; ++i) s += sred[i];
    atomicAdd(&out[n], s);
  }
}

// ---------------- 2a) grd -> normalized bf16 B[o][k] ----------------
__global__ __launch_bounds__(256)
void k_cvt_grd(const float* __restrict__ grd, const float* __restrict__ ssg,
               unsigned short* __restrict__ gbf) {
  int idx = blockIdx.x * 256 + threadIdx.x;       // < 16*32768
  int n = idx >> 15;
  float r = rinv_norm(ssg[n]);
  gbf[idx] = f2bf(grd[idx] * r);
}

// ---------------- 2b) sat -> normalized bf16 padded image ----------------
__global__ __launch_bounds__(256)
void k_cvt_sat(const float* __restrict__ sat, const float* __restrict__ sss,
               unsigned short* __restrict__ sbf) {
  int idx = blockIdx.x * 256 + threadIdx.x;       // < 16*32*128*128
  int w = idx & 127, h = (idx >> 7) & 127, c = (idx >> 14) & 31, n = idx >> 19;
  float v = 0.f;
  if (h >= 16 && h < 16 + HS && w >= 16 && w < 16 + WS) {
    v = sat[(((size_t)(n * CCH + c)) * HS + (h - 16)) * WS + (w - 16)] * rinv_norm(sss[n]);
  }
  sbf[idx] = f2bf(v);
}

// ---------------- 3a) fp32 energy image (raw, scaled later by rn^2) -------
__global__ __launch_bounds__(256)
void k_energy(const float* __restrict__ sat, float* __restrict__ E) {
  int idx = blockIdx.x * 256 + threadIdx.x;       // < 16*128*128
  int w = idx & 127, h = (idx >> 7) & 127, n = idx >> 14;
  float e = 0.f;
  if (h >= 16 && h < 16 + HS && w >= 16 && w < 16 + WS) {
    const float* p = sat + (((size_t)n * CCH) * HS + (h - 16)) * WS + (w - 16);
    #pragma unroll 4
    for (int c = 0; c < CCH; ++c) { float s = p[(size_t)c * HS * WS]; e += s * s; }
  }
  E[idx] = e;
}

// ---------------- 3b) 32-wide row sums ----------------
__global__ __launch_bounds__(256)
void k_rowsum(const float* __restrict__ E, float* __restrict__ R) {
  int idx = blockIdx.x * 256 + threadIdx.x;       // < 16*128*97
  int t = idx; int x = t % WO; t /= WO; int h = t % HP; int n = t / HP;
  const float* p = E + ((size_t)n * HP + h) * WP + x;
  float s = 0.f;
  #pragma unroll 8
  for (int j = 0; j < WK; ++j) s += p[j];
  R[idx] = s;
}

// ---------------- 3c) 32-tall column sums -> partial ----------------
__global__ __launch_bounds__(256)
void k_partial(const float* __restrict__ R, float* __restrict__ P, int total) {
  int idx = blockIdx.x * 256 + threadIdx.x;       // < 16*97*97
  if (idx >= total) return;
  int t = idx; int x = t % WO; t /= WO; int y = t % HO; int n = t / HO;
  const float* p = R + ((size_t)n * HP + y) * WO + x;
  float s = 0.f;
  #pragma unroll 8
  for (int i = 0; i < HK; ++i) s += p[(size_t)i * WO];
  P[idx] = s;
}

// ---------------- TDM descriptor issue: grd[0..15][k0..k0+KC) -> LDS ------
#if HAVE_TDM
static __device__ __forceinline__ void tdm_load_B(unsigned int lds_off,
                                                  const unsigned short* gbf, int k0) {
  uint64_t ga = (uint64_t)(uintptr_t)gbf + (uint64_t)k0 * 2u;
  tdm_u4 g0;
  g0.x = 1u;                                         // count=1 (valid user D#)
  g0.y = lds_off;                                    // lds_addr (bytes)
  g0.z = (unsigned int)ga;                           // global_addr[31:0]
  g0.w = (unsigned int)((ga >> 32) & 0x1FFFFFFu) | 0x80000000u; // addr[56:32] | type=2
  tdm_i8 g1;
  g1.s0 = 0x00010000;                                // data_size=2B, no mask/flags
  g1.s1 = (int)0x80000000u;                          // tensor_dim0=32768 (low16 in [63:48])
  g1.s2 = 16 << 16;                                  // tensor_dim1=16 (low16 in [95:80])
  g1.s3 = KC << 16;                                  // tile_dim0=1024 ([127:112])
  g1.s4 = 16;                                        // tile_dim1=16 ([143:128]); tile_dim2=0
  g1.s5 = 32768;                                     // tensor_dim0_stride low32
  g1.s6 = 0;                                         // stride0 hi / stride1 lo
  g1.s7 = 0;
  tdm_i4 z4; z4.x = 0; z4.y = 0; z4.z = 0; z4.w = 0; // groups 2/3 unused (2-D tile)
  tdm_i8 z8; z8.s0 = 0; z8.s1 = 0; z8.s2 = 0; z8.s3 = 0;
             z8.s4 = 0; z8.s5 = 0; z8.s6 = 0; z8.s7 = 0;
  // 6-arg toolchain variant: (g0 v4u, g1 v8i, g2 v4i, g3 v4i, v8i, cpol)
  __builtin_amdgcn_tensor_load_to_lds(g0, g1, z4, z4, z8, 0);
}
#endif

// ---------------- 4) WMMA GEMM + fused epilogue ----------------
// one wave = two vertically adjacent 16-position tiles (rows y0, y0+1):
// the A fragment of row (y0+1, i) equals row (y0, i+1) -> rolling register reuse,
// and both WMMAs share one B fragment. M=32 total, N=16, K=32768 per wave.
__global__ __launch_bounds__(256)
void k_gemm(const unsigned short* __restrict__ sbf,
            const unsigned short* __restrict__ gbf,  // B[o][k] bf16
            const float* __restrict__ ss,            // [0..15]=grd, [16..31]=sat
            const float* __restrict__ P,
            const int* __restrict__ test_method,
            float* __restrict__ tilemax) {
  __shared__ u32x4 ldsB[2][16 * KC / 8];             // double-buffered 2x32 KB

  const int tid  = threadIdx.x;
  const int lane = tid & 31;
  const int lo16 = lane & 15;
  const int hi   = lane >> 4;
  const bool w0  = (tid < 32);

  int wid = blockIdx.x * 8 + (tid >> 5);
  const bool wactive = wid < TOT2;
  const int wc  = wactive ? wid : (TOT2 - 1);
  const int n   = wc / T2;
  const int rem = wc % T2;
  const int y0  = (rem / TXN) * 2;                   // 0..96 even
  const int x0  = (rem % TXN) * 16;

  const unsigned short* satn = sbf + (size_t)n * (CCH * HP * WP);
  const int acol = x0 + lo16 + 8 * hi;               // per-lane base column

  v8f acc0 = (v8f)(0.0f);
  v8f acc1 = (v8f)(0.0f);

  // ---- preload chunk 0 of the B panel ----
#if HAVE_TDM
  if (w0) {
    tdm_load_B((unsigned int)(uintptr_t)&ldsB[0][0], gbf, 0);
    __builtin_amdgcn_s_wait_tensorcnt(0);
  }
#else
  #pragma unroll
  for (int r = 0; r < 8; ++r) {
    int q = r * 256 + tid;
    ldsB[0][q] = ((const u32x4*)gbf)[(size_t)(q >> 7) * (KTOT / 8) + (q & 127)];
  }
#endif
  __syncthreads();

  for (int c0 = 0; c0 < CCH; ++c0) {                 // one channel per chunk
    const int buf = c0 & 1;
    // ---- kick off DMA of the next chunk into the other buffer ----
#if HAVE_TDM
    if (w0 && (c0 + 1) < CCH)
      tdm_load_B((unsigned int)(uintptr_t)&ldsB[buf ^ 1][0], gbf, (c0 + 1) * KC);
#else
    if ((c0 + 1) < CCH) {
      #pragma unroll
      for (int r = 0; r < 8; ++r) {
        int q = r * 256 + tid;
        ldsB[buf ^ 1][q] =
            ((const u32x4*)gbf)[(size_t)(q >> 7) * (KTOT / 8) + ((c0 + 1) * KC / 8) + (q & 127)];
      }
    }
#endif

    const unsigned short* rowbase = satn + ((size_t)c0 * HP + y0) * WP + acol;
    if ((c0 + 1) < CCH)                              // gfx1250 global_prefetch_b8
      __builtin_prefetch(rowbase + (size_t)HP * WP, 0, 0);

    // rolling A reuse across the 32 rows of this channel
    Frag32 cur;
    cur.lo = ldg_u4_unaligned(rowbase);
    cur.hi = ldg_u4_unaligned(rowbase + 16);
    #pragma unroll 4
    for (int i = 0; i < 32; ++i) {
      const unsigned short* rnx = rowbase + (size_t)(i + 1) * WP;
      Frag32 nxt;
      nxt.lo = ldg_u4_unaligned(rnx);
      nxt.hi = ldg_u4_unaligned(rnx + 16);
      const int b8 = lo16 * (KC / 8) + i * 4 + hi;
      Frag32 fb; fb.lo = ldsB[buf][b8]; fb.hi = ldsB[buf][b8 + 2];
      v16bf bv = __builtin_bit_cast(v16bf, fb);
      acc0 = __builtin_amdgcn_wmma_f32_16x16x32_bf16(
          false, __builtin_bit_cast(v16bf, cur), false, bv, (short)0, acc0, false, false);
      acc1 = __builtin_amdgcn_wmma_f32_16x16x32_bf16(
          false, __builtin_bit_cast(v16bf, nxt), false, bv, (short)0, acc1, false, false);
      cur = nxt;
    }

#if HAVE_TDM
    if (w0 && (c0 + 1) < CCH) __builtin_amdgcn_s_wait_tensorcnt(0);
#endif
    __syncthreads();                                  // next-buffer data visible / consumed
  }

  // ---- fused epilogue: corr / max(partial, eps), pair max ----
  const float rn  = rinv_norm(ss[16 + n]);
  const float rn2 = rn * rn;
  const int   tm  = *test_method;
  const float* Pn0 = P + ((size_t)n * HO + y0) * WO;
  const bool  row1 = (y0 + 1) < HO;
  float m = -__builtin_huge_valf();
  #pragma unroll
  for (int v = 0; v < 8; ++v) {
    const int x = x0 + v + 8 * hi;                   // D layout: lane row M = v + 8*(lane>=16)
    if (x < WO) {
      float p0 = Pn0[x] * rn2;
      float d0 = (tm == 0) ? fmaxf(sqrtf(p0), EPSV) : fmaxf(p0, EPSV);
      m = fmaxf(m, acc0[v] / d0);
      if (row1) {
        float p1 = Pn0[WO + x] * rn2;
        float d1 = (tm == 0) ? fmaxf(sqrtf(p1), EPSV) : fmaxf(p1, EPSV);
        m = fmaxf(m, acc1[v] / d1);
      }
    }
  }
  m = fmaxf(m, __shfl_xor(m, 16, 32));               // half-lanes share column o = lane&15
  if (wactive && hi == 0) tilemax[(size_t)wc * 16 + lo16] = m;
}

// ---------------- 5) final reduction over tile-pairs ----------------
__global__ __launch_bounds__(128)
void k_redmax(const float* __restrict__ tilemax, float* __restrict__ out) {
  const int pair = blockIdx.x;                       // n*16 + o
  const int n = pair >> 4, o = pair & 15;
  float m = -__builtin_huge_valf();
  for (int t = threadIdx.x; t < T2; t += 128)
    m = fmaxf(m, tilemax[(size_t)(n * T2 + t) * 16 + o]);
  #pragma unroll
  for (int off = 16; off; off >>= 1) m = fmaxf(m, __shfl_down(m, off, 32));
  __shared__ float sm[4];
  if ((threadIdx.x & 31) == 0) sm[threadIdx.x >> 5] = m;
  __syncthreads();
  if (threadIdx.x == 0)
    out[pair] = fmaxf(fmaxf(sm[0], sm[1]), fmaxf(sm[2], sm[3]));
}

// ---------------- host launcher ----------------
extern "C" void kernel_launch(void* const* d_in, const int* in_sizes, int n_in,
                              void* d_out, int out_size, void* d_ws, size_t ws_size,
                              hipStream_t stream) {
  const float* grd = (const float*)d_in[0];          // (16,32,32,32)
  const float* sat = (const float*)d_in[1];          // (16,32,96,96)
  const int*   tm  = (const int*)d_in[2];
  float* out = (float*)d_out;                        // (16,16)

  // workspace carve-up (all 16B-aligned by construction)
  char* ws = (char*)d_ws;
  unsigned short* sbf = (unsigned short*)(ws);       // 16 MB padded bf16 sat
  unsigned short* gbf = (unsigned short*)(ws + 16777216); // 1 MB bf16 grd
  float* ss   = (float*)(ws + 17825792);   // [0..15] grd sumsq, [16..31] sat sumsq
  float* E    = (float*)(ws + 17825920);   // 16*128*128
  float* R    = (float*)(ws + 18874496);   // 16*128*97
  float* P    = (float*)(ws + 19669120);   // 16*97*97
  float* tmax = (float*)(ws + 20271360);   // 5488*16

  (void)hipMemsetAsync(ss, 0, 32 * sizeof(float), stream);

  // per-sample sums of squares
  k_sumsq<<<NB * (KTOT / 256),          256, 0, stream>>>(grd, ss,      KTOT / 256,          KTOT);
  k_sumsq<<<NB * (CCH * HS * WS / 256), 256, 0, stream>>>(sat, ss + 16, CCH * HS * WS / 256, CCH * HS * WS);

  // normalize + convert to bf16
  k_cvt_grd<<<NB * KTOT / 256,          256, 0, stream>>>(grd, ss, gbf);
  k_cvt_sat<<<NB * CCH * HP * WP / 256, 256, 0, stream>>>(sat, ss + 16, sbf);

  // denominator: energy image -> separable 32x32 window sums
  k_energy <<<NB * HP * WP / 256,         256, 0, stream>>>(sat, E);
  k_rowsum <<<NB * HP * WO / 256,         256, 0, stream>>>(E, R);
  k_partial<<<(NB * HO * WO + 255) / 256, 256, 0, stream>>>(R, P, NB * HO * WO);

  // main WMMA correlation + fused normalize/max epilogue
  k_gemm<<<(TOT2 + 7) / 8, 256, 0, stream>>>(sbf, gbf, ss, P, tm, tmax);

  // final (16,16) output
  k_redmax<<<NB * NB, 128, 0, stream>>>(tmax, out);
}
